// AttentionFuser_90735479096012
// MI455X (gfx1250) — compile-verified, matrix operand order checked by reading, not run on hardware
//
#include <hip/hip_runtime.h>
#include <cstdint>

// ---------------------------------------------------------------------------
// Types for CDNA5 WMMA (gfx1250, wave32)
// ---------------------------------------------------------------------------
typedef __bf16 bf16_t;
typedef __attribute__((ext_vector_type(16))) __bf16 v16bf;
typedef __attribute__((ext_vector_type(8)))  float  v8f;
typedef __attribute__((ext_vector_type(4)))  int    v4i;

union FragU { v16bf v; uint4 u[2]; };

__device__ inline v8f wmma_bf16(v16bf a, v16bf b, v8f c) {
  // (neg_a, A, neg_b, B, c_mod, C, reuse_a, reuse_b)
  return __builtin_amdgcn_wmma_f32_16x16x32_bf16(false, a, false, b, (short)0, c,
                                                 false, false);
}

// ---------------------------------------------------------------------------
// CDNA5 async global->LDS copies (ASYNCcnt / s_wait_asynccnt), guarded so the
// file still compiles (with plain LDS stores) if the builtins are absent.
// Builtin signature (probed): (v4i* AS1, v4i* AS3, imm offset, imm cpol)
// ---------------------------------------------------------------------------
#if defined(__has_builtin)
#  if __has_builtin(__builtin_amdgcn_global_load_async_to_lds_b128) && \
      __has_builtin(__builtin_amdgcn_s_wait_asynccnt)
#    define HAVE_ASYNC_LDS 1
#  endif
#endif
#ifndef HAVE_ASYNC_LDS
#  define HAVE_ASYNC_LDS 0
#endif

typedef __attribute__((address_space(1))) v4i gas_v4i;
typedef __attribute__((address_space(3))) v4i lds_v4i;

__device__ inline void async_copy_b128(const bf16_t* g, bf16_t* l) {
#if HAVE_ASYNC_LDS
  __builtin_amdgcn_global_load_async_to_lds_b128(
      (gas_v4i*)(v4i*)(void*)g, (lds_v4i*)(v4i*)(void*)l, 0, 0);
#else
  *(uint4*)l = *(const uint4*)g;
#endif
}

template <int N> __device__ inline void wait_async_le() {
#if HAVE_ASYNC_LDS
  __builtin_amdgcn_s_wait_asynccnt((unsigned short)N);
#endif
}

#define TM 128   // workgroup M tile (8 waves x 16 rows)
#define TN 64    // workgroup N tile (4 wmma tiles per wave)
#define TK 32    // K step per wmma (bf16)
#define TKP 40   // padded LDS row stride for transposed B tile (80B, 16B-aligned)

// ---------------------------------------------------------------------------
// fp32 -> bf16 conversion (vectorized x4)
// ---------------------------------------------------------------------------
__global__ __launch_bounds__(256)
void cvt_bf16_kernel(const float* __restrict__ src, bf16_t* __restrict__ dst,
                     long long n) {
  long long i = ((long long)blockIdx.x * 256 + threadIdx.x) * 4;
  if (i >= n) return;
  float4 f = *(const float4*)(src + i);
  union { bf16_t e[4]; uint2 u; } o;
  o.e[0] = (bf16_t)f.x; o.e[1] = (bf16_t)f.y;
  o.e[2] = (bf16_t)f.z; o.e[3] = (bf16_t)f.w;
  *(uint2*)(dst + i) = o.u;
}

// ---------------------------------------------------------------------------
// Copy a compact (rows, D) f32 tensor into a strided slice of the output
// ---------------------------------------------------------------------------
__global__ __launch_bounds__(256)
void copy_slice_kernel(const float4* __restrict__ in, float4* __restrict__ out,
                       int ncol4, int ldo4, long long total4) {
  long long i = (long long)blockIdx.x * 256 + threadIdx.x;
  if (i >= total4) return;
  long long row = i / ncol4;
  int c4 = (int)(i - row * ncol4);
  out[row * ldo4 + c4] = in[i];
}

// ---------------------------------------------------------------------------
// GEMM NT:  C[m,n] = sum_k A[m,k] * B[n,k]     (A: MxK, B: NxK, bf16)
// Double-buffered async global->LDS pipeline. Optional f32 and/or bf16 output.
// Batched via blockIdx.z. All dims divisible by the tile sizes.
// ---------------------------------------------------------------------------
__global__ __launch_bounds__(256)
void gemm_nt_bf16(const bf16_t* __restrict__ A, long long strideA,
                  const bf16_t* __restrict__ Bm, long long strideB,
                  float* __restrict__ Cf, int ldc, long long strideCf,
                  bf16_t* __restrict__ Cb, int ldcb, long long strideCb,
                  int K) {
  __shared__ __align__(16) bf16_t sA[2][TM * TK];
  __shared__ __align__(16) bf16_t sB[2][TN * TK];

  const int b = blockIdx.z;
  const bf16_t* Ab = A  + (long long)b * strideA;
  const bf16_t* Bb = Bm + (long long)b * strideB;

  const int tid  = threadIdx.x;
  const int wave = tid >> 5;
  const int lane = tid & 31;
  const int half = lane >> 4;
  const int lr   = lane & 15;
  const long long m0 = (long long)blockIdx.y * TM;
  const long long n0 = (long long)blockIdx.x * TN;

  v8f acc[4];
#pragma unroll
  for (int t = 0; t < 4; ++t)
#pragma unroll
    for (int j = 0; j < 8; ++j) acc[t][j] = 0.0f;

  // per-thread staging addresses (3 async b128 copies per tile)
  auto stage = [&](int buf, int k0) {
#pragma unroll
    for (int i = 0; i < 2; ++i) {
      int idx = tid + i * 256;
      int row = idx >> 2;
      int kv  = (idx & 3) * 8;
      async_copy_b128(&Ab[(m0 + row) * (long long)K + k0 + kv],
                      &sA[buf][row * TK + kv]);
    }
    {
      int row = tid >> 2;
      int kv  = (tid & 3) * 8;
      async_copy_b128(&Bb[(n0 + row) * (long long)K + k0 + kv],
                      &sB[buf][row * TK + kv]);
    }
  };

  stage(0, 0);
  int buf = 0;
  for (int k0 = 0; k0 < K; k0 += TK, buf ^= 1) {
    const bool more = (k0 + TK) < K;
    // prefetch next tile into the other buffer while this one computes
    if (more) {
      stage(buf ^ 1, k0 + TK);
      wait_async_le<3>();   // drain this tile's 3 copies (in-order completion)
    } else {
      wait_async_le<0>();
    }
    __syncthreads();

    // A fragment: lanes 0-15 -> K 0..7 / 16..23 ; lanes 16-31 -> K 8..15 / 24..31
    FragU af;
    int arow = wave * 16 + lr;
    af.u[0] = *(const uint4*)(&sA[buf][arow * TK + half * 8]);
    af.u[1] = *(const uint4*)(&sA[buf][arow * TK + half * 8 + 16]);
#pragma unroll
    for (int t = 0; t < 4; ++t) {
      FragU bfr;
      int brow = t * 16 + lr;
      bfr.u[0] = *(const uint4*)(&sB[buf][brow * TK + half * 8]);
      bfr.u[1] = *(const uint4*)(&sB[buf][brow * TK + half * 8 + 16]);
      acc[t] = wmma_bf16(af.v, bfr.v, acc[t]);
    }
    __syncthreads();   // protect buffer reuse by the next stage()
  }

  // C layout: element j of v8f -> row = j + half*8, col = lr (per 16x16 tile)
  const long long mrow = m0 + wave * 16;
#pragma unroll
  for (int t = 0; t < 4; ++t) {
    long long col = n0 + t * 16 + lr;
#pragma unroll
    for (int j = 0; j < 8; ++j) {
      long long row = mrow + j + half * 8;
      float v = acc[t][j];
      if (Cf) Cf[(long long)b * strideCf + row * ldc + col] = v;
      if (Cb) Cb[(long long)b * strideCb + row * ldcb + col] = (bf16_t)v;
    }
  }
}

// ---------------------------------------------------------------------------
// GEMM NN:  C[m,n] = sum_k A[m,k] * B[k,n]   (A: MxK bf16, B: KxN bf16)
// A tile staged via async copies; B tile transposed through VGPRs into padded
// LDS. f32 output with optional accumulate + up to two f32 residual adds,
// optional bf16 side output. Batched via blockIdx.z.
// ---------------------------------------------------------------------------
__global__ __launch_bounds__(256)
void gemm_nn_bf16(const bf16_t* __restrict__ A, long long strideA,
                  const bf16_t* __restrict__ Bm, long long strideB,
                  float* __restrict__ Cf, int ldc, long long strideCf,
                  const float* __restrict__ R0, const float* __restrict__ R1,
                  int ldr, long long strideR,
                  int accumulate,
                  bf16_t* __restrict__ Cb, int ldcb, long long strideCb,
                  int K, int N) {
  __shared__ __align__(16) bf16_t sA[TM * TK];
  __shared__ __align__(16) bf16_t sBt[TN * TKP];   // transposed: [n][k], padded

  const int b = blockIdx.z;
  const bf16_t* Ab = A  + (long long)b * strideA;
  const bf16_t* Bb = Bm + (long long)b * strideB;

  const int tid  = threadIdx.x;
  const int wave = tid >> 5;
  const int lane = tid & 31;
  const int half = lane >> 4;
  const int lr   = lane & 15;
  const long long m0 = (long long)blockIdx.y * TM;
  const long long n0 = (long long)blockIdx.x * TN;

  const int kk = tid >> 3;        // 0..31  (B staging row)
  const int nn = (tid & 7) * 8;   // 0..56  (B staging col group)

  v8f acc[4];
#pragma unroll
  for (int t = 0; t < 4; ++t)
#pragma unroll
    for (int j = 0; j < 8; ++j) acc[t][j] = 0.0f;

  for (int k0 = 0; k0 < K; k0 += TK) {
    // stage A tile: 128x32 via async copies
#pragma unroll
    for (int i = 0; i < 2; ++i) {
      int idx = tid + i * 256;
      int row = idx >> 2;
      int kv  = (idx & 3) * 8;
      async_copy_b128(&Ab[(m0 + row) * (long long)K + k0 + kv],
                      &sA[row * TK + kv]);
    }
    // stage B tile 32(K) x 64(N), transposed into sBt[n][k]
    {
      union { uint4 u; bf16_t e[8]; } r;
      r.u = *(const uint4*)(&Bb[(long long)(k0 + kk) * N + n0 + nn]);
#pragma unroll
      for (int e = 0; e < 8; ++e) sBt[(nn + e) * TKP + kk] = r.e[e];
      if (k0 + TK < K)  // hint the next B tile toward the caches
        __builtin_prefetch(&Bb[(long long)(k0 + TK + kk) * N + n0 + nn], 0, 0);
    }
    wait_async_le<0>();
    __syncthreads();

    FragU af;
    int arow = wave * 16 + lr;
    af.u[0] = *(const uint4*)(&sA[arow * TK + half * 8]);
    af.u[1] = *(const uint4*)(&sA[arow * TK + half * 8 + 16]);
#pragma unroll
    for (int t = 0; t < 4; ++t) {
      FragU bfr;
      int brow = t * 16 + lr;
      bfr.u[0] = *(const uint4*)(&sBt[brow * TKP + half * 8]);
      bfr.u[1] = *(const uint4*)(&sBt[brow * TKP + half * 8 + 16]);
      acc[t] = wmma_bf16(af.v, bfr.v, acc[t]);
    }
    __syncthreads();
  }

  const long long mrow = m0 + wave * 16;
#pragma unroll
  for (int t = 0; t < 4; ++t) {
    long long col = n0 + t * 16 + lr;
#pragma unroll
    for (int j = 0; j < 8; ++j) {
      long long row = mrow + j + half * 8;
      float v = acc[t][j];
      long long co = (long long)b * strideCf + row * ldc + col;
      if (accumulate) v += Cf[co];
      if (R0) v += R0[(long long)b * strideR + row * ldr + col];
      if (R1) v += R1[(long long)b * strideR + row * ldr + col];
      Cf[co] = v;
      if (Cb) Cb[(long long)b * strideCb + row * ldcb + col] = (bf16_t)v;
    }
  }
}

// ---------------------------------------------------------------------------
// Row softmax over length-1024 rows: P = softmax(S * scale), f32 -> bf16
// One 256-thread workgroup (8 wave32) per row; 4 elements per thread.
// ---------------------------------------------------------------------------
__global__ __launch_bounds__(256)
void softmax_rows_kernel(const float* __restrict__ S, bf16_t* __restrict__ P,
                         float scale) {
  const long long row = blockIdx.x;
  const float* s = S + row * 1024;
  bf16_t* p = P + row * 1024;
  const int tid = threadIdx.x;

  float vals[4];
  float mx = -3.4e38f;
#pragma unroll
  for (int i = 0; i < 4; ++i) {
    vals[i] = s[tid + i * 256] * scale;
    mx = fmaxf(mx, vals[i]);
  }
#pragma unroll
  for (int o = 16; o > 0; o >>= 1) mx = fmaxf(mx, __shfl_xor(mx, o, 32));

  __shared__ float red[8];
  if ((tid & 31) == 0) red[tid >> 5] = mx;
  __syncthreads();
  float m = red[0];
#pragma unroll
  for (int i = 1; i < 8; ++i) m = fmaxf(m, red[i]);
  __syncthreads();

  float sum = 0.0f;
#pragma unroll
  for (int i = 0; i < 4; ++i) {
    vals[i] = __expf(vals[i] - m);
    sum += vals[i];
  }
#pragma unroll
  for (int o = 16; o > 0; o >>= 1) sum += __shfl_xor(sum, o, 32);
  if ((tid & 31) == 0) red[tid >> 5] = sum;
  __syncthreads();
  float tot = red[0];
#pragma unroll
  for (int i = 1; i < 8; ++i) tot += red[i];
  float inv = 1.0f / tot;
#pragma unroll
  for (int i = 0; i < 4; ++i) p[tid + i * 256] = (bf16_t)(vals[i] * inv);
}

// ---------------------------------------------------------------------------
// Host orchestration
// ---------------------------------------------------------------------------
extern "C" void kernel_launch(void* const* d_in, const int* in_sizes, int n_in,
                              void* d_out, int out_size, void* d_ws, size_t ws_size,
                              hipStream_t stream) {
  (void)in_sizes; (void)n_in; (void)out_size; (void)ws_size;

  const int  Bb = 16, L = 1024, D = 512;
  const long long BL  = (long long)Bb * L;        // 16384
  const long long BLD = BL * D;                   // 8,388,608
  const long long DD  = (long long)D * D;         // 262,144
  const long long BLL = (long long)Bb * L * L;    // 16,777,216
  const float SCALE = 0.044194173824159216f;      // 1/sqrt(512)

  const float* txt_f = (const float*)d_in[0];
  const float* au_f  = (const float*)d_in[1];
  const float* vi_f  = (const float*)d_in[2];
  float* out = (float*)d_out;

  // ---- workspace carve-up (bf16 region then f32 scores) ----
  bf16_t* wsb = (bf16_t*)d_ws;
  bf16_t* txt_b = wsb;             wsb += BLD;
  bf16_t* au_b  = wsb;             wsb += BLD;
  bf16_t* vi_b  = wsb;             wsb += BLD;
  bf16_t* w_b   = wsb;             wsb += 21 * DD;   // 21 weights
  bf16_t* p0 = wsb; wsb += BLD;
  bf16_t* p1 = wsb; wsb += BLD;
  bf16_t* p2 = wsb; wsb += BLD;
  bf16_t* p3 = wsb; wsb += BLD;
  bf16_t* p4 = wsb; wsb += BLD;
  bf16_t* p5 = wsb; wsb += BLD;
  bf16_t* av_b = wsb; wsb += BLD;
  float*  scores = (float*)wsb;
  bf16_t* probs  = (bf16_t*)(scores + BLL);

  dim3 blk(256);

  auto cvt = [&](const float* s, bf16_t* d, long long n) {
    long long thr = n / 4;
    cvt_bf16_kernel<<<dim3((unsigned)((thr + 255) / 256)), blk, 0, stream>>>(s, d, n);
  };

  // NT gemm: C = A * B^T  (B stored N x K)
  auto gemm_nt = [&](const bf16_t* A, long long sA, const bf16_t* Bm, long long sB,
                     float* Cf, int ldc, long long sCf,
                     bf16_t* Cb, int ldcb, long long sCb,
                     int M, int N, int K, int batch) {
    dim3 grid((unsigned)(N / TN), (unsigned)(M / TM), (unsigned)batch);
    gemm_nt_bf16<<<grid, blk, 0, stream>>>(A, sA, Bm, sB, Cf, ldc, sCf,
                                           Cb, ldcb, sCb, K);
  };

  // NN gemm: C = A * B (B stored K x N), residuals + accumulate + bf16 out
  auto gemm_nn = [&](const bf16_t* A, long long sA, const bf16_t* Bm, long long sB,
                     float* Cf, int ldc, long long sCf,
                     const float* R0, const float* R1, int ldr, long long sR,
                     int accum, bf16_t* Cb, int ldcb, long long sCb,
                     int M, int N, int K, int batch) {
    dim3 grid((unsigned)(N / TN), (unsigned)(M / TM), (unsigned)batch);
    gemm_nn_bf16<<<grid, blk, 0, stream>>>(A, sA, Bm, sB, Cf, ldc, sCf,
                                           R0, R1, ldr, sR, accum,
                                           Cb, ldcb, sCb, K, N);
  };

  auto softmax_all = [&]() {
    softmax_rows_kernel<<<dim3((unsigned)BL), blk, 0, stream>>>(scores, probs, SCALE);
  };

  // ---- 1) conversions ----
  cvt(txt_f, txt_b, BLD);
  cvt(au_f,  au_b,  BLD);
  cvt(vi_f,  vi_b,  BLD);
  for (int i = 0; i < 21; ++i)
    cvt((const float*)d_in[3 + i], w_b + (long long)i * DD, DD);

  // ---- 2) passthrough copies: txt -> col 0, au -> col D, vi -> col 2D ----
  const int ld7 = 7 * D;
  {
    long long total4 = BLD / 4;
    unsigned g = (unsigned)((total4 + 255) / 256);
    copy_slice_kernel<<<g, blk, 0, stream>>>((const float4*)txt_f, (float4*)(out),
                                             D / 4, ld7 / 4, total4);
    copy_slice_kernel<<<g, blk, 0, stream>>>((const float4*)au_f,
                                             (float4*)(out + D), D / 4, ld7 / 4, total4);
    copy_slice_kernel<<<g, blk, 0, stream>>>((const float4*)vi_f,
                                             (float4*)(out + 2 * D), D / 4, ld7 / 4, total4);
  }

  // ---- 3) symmetric attention blocks ----
  // weight order per block: kx, qx, vx, ky, qy, vy
  auto wb = [&](int i) { return w_b + (long long)i * DD; };

  auto sym_block = [&](const bf16_t* m1b, const bf16_t* m2b,
                       const float* m1f, const float* m2f,
                       int wbase, int coloff, bf16_t* bf_out) {
    // projections (output bf16 only)
    gemm_nt(m1b, 0, wb(wbase + 0), 0, nullptr, 0, 0, p0, D, 0, (int)BL, D, D, 1); // k1
    gemm_nt(m2b, 0, wb(wbase + 4), 0, nullptr, 0, 0, p1, D, 0, (int)BL, D, D, 1); // q2
    gemm_nt(m1b, 0, wb(wbase + 2), 0, nullptr, 0, 0, p2, D, 0, (int)BL, D, D, 1); // v1
    gemm_nt(m2b, 0, wb(wbase + 3), 0, nullptr, 0, 0, p3, D, 0, (int)BL, D, D, 1); // k2
    gemm_nt(m1b, 0, wb(wbase + 1), 0, nullptr, 0, 0, p4, D, 0, (int)BL, D, D, 1); // q1
    gemm_nt(m2b, 0, wb(wbase + 5), 0, nullptr, 0, 0, p5, D, 0, (int)BL, D, D, 1); // v2

    // s1 = q2 k1^T  (batched over b)
    gemm_nt(p1, (long long)L * D, p0, (long long)L * D,
            scores, L, (long long)L * L, nullptr, 0, 0, L, L, D, Bb);
    softmax_all();
    // o  = s1 v1 + m1 + m2  -> out slice
    gemm_nn(probs, (long long)L * L, p2, (long long)L * D,
            out + coloff, ld7, (long long)L * ld7,
            m1f, m2f, D, (long long)L * D,
            0, nullptr, 0, 0, L, D, L, Bb);
    // s2 = q1 k2^T
    gemm_nt(p4, (long long)L * D, p3, (long long)L * D,
            scores, L, (long long)L * L, nullptr, 0, 0, L, L, D, Bb);
    softmax_all();
    // out slice += s2 v2   (and optional bf16 copy of the final slice)
    gemm_nn(probs, (long long)L * L, p5, (long long)L * D,
            out + coloff, ld7, (long long)L * ld7,
            nullptr, nullptr, 0, 0,
            1, bf_out, D, (long long)L * D, L, D, L, Bb);
  };

  sym_block(txt_b, au_b, txt_f, au_f,  0, 3 * D, nullptr);  // ta = o1+o2
  sym_block(vi_b,  au_b, vi_f,  au_f,  6, 5 * D, av_b);     // av = o1+o2 (keep bf16)
  sym_block(txt_b, vi_b, txt_f, vi_f, 12, 4 * D, nullptr);  // tv = o1+o2

  // ---- 4) cross attention: tav = attn(x=txt, queries=av) ----
  gemm_nt(txt_b, 0, wb(18), 0, nullptr, 0, 0, p0, D, 0, (int)BL, D, D, 1); // k
  gemm_nt(av_b,  0, wb(19), 0, nullptr, 0, 0, p1, D, 0, (int)BL, D, D, 1); // q
  gemm_nt(txt_b, 0, wb(20), 0, nullptr, 0, 0, p2, D, 0, (int)BL, D, D, 1); // v
  gemm_nt(p1, (long long)L * D, p0, (long long)L * D,
          scores, L, (long long)L * L, nullptr, 0, 0, L, L, D, Bb);
  softmax_all();
  gemm_nn(probs, (long long)L * L, p2, (long long)L * D,
          out + 6 * D, ld7, (long long)L * ld7,
          nullptr, nullptr, 0, 0,
          0, nullptr, 0, 0, L, D, L, Bb);
}
